// PretrainedDetector_81939386073145
// MI455X (gfx1250) — matrix operand order repaired and checked by reading, not compiled, not run on hardware
//
#include <hip/hip_runtime.h>
#include <hip/hip_bf16.h>

// Problem dims
#define B_   8
#define L_   4096
#define DM   256
#define DS   16
#define DI   512
#define DTR  16
#define M_   (B_*L_)        // 32768 tokens
#define SCHUNK 64           // scan chunk length
#define NCH  (L_/SCHUNK)    // 64 chunks

typedef __attribute__((ext_vector_type(16))) __bf16 v16bf;
typedef __attribute__((ext_vector_type(8)))  __bf16 v8bf;
typedef __attribute__((ext_vector_type(8)))  float  v8f;

// ---------------------------------------------------------------------------
// K0: convert in_w (1024x256) and xp_w (48x512) fp32 -> bf16
// ---------------------------------------------------------------------------
__global__ void k_convert_weights(const float* __restrict__ in_w,
                                  const float* __restrict__ xp_w,
                                  __bf16* __restrict__ in_w_bf,
                                  __bf16* __restrict__ xp_w_bf) {
  int i = blockIdx.x * blockDim.x + threadIdx.x;
  const int n1 = 1024 * 256;
  if (i < n1) {
    in_w_bf[i] = (__bf16)in_w[i];
  } else {
    int j = i - n1;
    if (j < 48 * 512) xp_w_bf[j] = (__bf16)xp_w[j];
  }
}

// ---------------------------------------------------------------------------
// K1: h = x @ ip_w.T + ip_b  (K=4), output bf16 [M_, 256]
// ---------------------------------------------------------------------------
__global__ void k_inproj(const float* __restrict__ x,
                         const float* __restrict__ ip_w,
                         const float* __restrict__ ip_b,
                         __bf16* __restrict__ hbf) {
  int i = blockIdx.x * blockDim.x + threadIdx.x;  // m*256 + j
  int m = i >> 8, j = i & 255;
  const float* xr = x + m * 4;
  const float* wr = ip_w + j * 4;
  float acc = ip_b[j] + xr[0]*wr[0] + xr[1]*wr[1] + xr[2]*wr[2] + xr[3]*wr[3];
  hbf[i] = (__bf16)acc;
}

// ---------------------------------------------------------------------------
// K2: xz = h @ in_w.T via WMMA bf16.  M=32768, N=1024, K=256.
//  - Block = 8 waves = one 16-row M tile covering all 1024 N columns.
//  - A tile (16x256 bf16, 8KB) staged in LDS once, shared by all waves.
//  - Each wave: 8 N-tiles as 2 groups of 4 (one A fragment feeds 4 WMMAs).
//  - cols <512 -> xi (fp32); cols >=512 -> z kept only at t=L-1.
// ---------------------------------------------------------------------------
__global__ void k_gemm_xz(const __bf16* __restrict__ hbf,
                          const __bf16* __restrict__ wbf,
                          float* __restrict__ xi,
                          float* __restrict__ z_last) {
  const int K = 256;
  __shared__ __bf16 As[16 * 256];           // 8 KB

  int m0 = blockIdx.x << 4;
  // Cooperative stage of A tile: 256 threads x 32B
  {
    int r  = threadIdx.x >> 4;              // 0..15
    int ck = (threadIdx.x & 15) << 4;       // 0..240
    *(v16bf*)(As + r * K + ck) = *(const v16bf*)(hbf + (m0 + r) * K + ck);
  }
  __syncthreads();

  int wave = threadIdx.x >> 5;
  int lane = threadIdx.x & 31;
  int aoff  = (lane >> 4) << 4;             // 0 or 16
  int arow  = lane & 15;
  int bkoff = (lane >> 4) << 4;             // 0 or 16
  int brow  = lane & 15;
  const __bf16* Alds = As + arow * K;

#pragma unroll
  for (int g = 0; g < 2; ++g) {
    int ntbase = wave * 8 + g * 4;          // first of 4 N-tiles
    const __bf16* bp[4];
#pragma unroll
    for (int t = 0; t < 4; ++t)
      bp[t] = wbf + ((ntbase + t) * 16 + brow) * K + bkoff;

    v8f acc[4];
#pragma unroll
    for (int t = 0; t < 4; ++t) acc[t] = (v8f){};

#pragma unroll
    for (int kk = 0; kk < 8; ++kk) {
      int k0 = kk * 32;
      v8bf alo = *(const v8bf*)(Alds + k0 + aoff);
      v8bf ahi = *(const v8bf*)(Alds + k0 + 8 + aoff);
      v16bf a = __builtin_shufflevector(alo, ahi,
                  0,1,2,3,4,5,6,7,8,9,10,11,12,13,14,15);
#pragma unroll
      for (int t = 0; t < 4; ++t) {
        v16bf b = *(const v16bf*)(bp[t] + k0);
        acc[t] = __builtin_amdgcn_wmma_f32_16x16x32_bf16(
                    false, a, false, b, (short)0, acc[t], false, false);
      }
    }

    int rbase = m0 + ((lane >> 4) << 3);
#pragma unroll
    for (int t = 0; t < 4; ++t) {
      int col = (ntbase + t) * 16 + (lane & 15);
#pragma unroll
      for (int v = 0; v < 8; ++v) {
        int r = rbase + v;
        if (col < DI) {
          xi[r * DI + col] = acc[t][v];
        } else if ((r & (L_ - 1)) == (L_ - 1)) {
          z_last[(r >> 12) * DI + (col - DI)] = acc[t][v];  // r>>12 == batch
        }
      }
    }
  }
}

// ---------------------------------------------------------------------------
// K3: causal depthwise conv (k=4) + SiLU -> xc bf16 [M_, 512]
// ---------------------------------------------------------------------------
__global__ void k_conv_silu(const float* __restrict__ xi,
                            const float* __restrict__ conv_w,
                            const float* __restrict__ conv_b,
                            __bf16* __restrict__ xc_bf) {
  int i = blockIdx.x * blockDim.x + threadIdx.x;  // (b*L+t)*512 + d
  int d = i & 511;
  int m = i >> 9;
  int t = m & (L_ - 1);
  const float* w = conv_w + d * 4;
  float acc = conv_b[d];
#pragma unroll
  for (int j = 0; j < 4; ++j) {
    int tt = t - 3 + j;
    if (tt >= 0) acc += xi[(m + tt - t) * DI + d] * w[j];
  }
  float s = acc / (1.0f + __expf(-acc));
  xc_bf[i] = (__bf16)s;
}

// ---------------------------------------------------------------------------
// K4: x_dbl = xc @ xp_w.T via WMMA bf16.  M=32768, N=48, K=512. fp32 out.
// ---------------------------------------------------------------------------
__global__ void k_gemm_xdbl(const __bf16* __restrict__ xc_bf,
                            const __bf16* __restrict__ xpw_bf,
                            float* __restrict__ xdbl) {
  const int K = DI;
  int lane  = threadIdx.x & 31;
  int gwave = (blockIdx.x * blockDim.x + threadIdx.x) >> 5;
  int mt = gwave / 3, nt = gwave % 3;
  int m0 = mt << 4, n0 = nt << 4;

  const __bf16* abase = xc_bf + (m0 + (lane & 15)) * K;
  int aoff  = (lane >> 4) << 4;
  const __bf16* bbase = xpw_bf + (n0 + (lane & 15)) * K;
  int bkoff = (lane >> 4) << 4;

  v8f c = {};
#pragma unroll
  for (int k0 = 0; k0 < K; k0 += 32) {
    v8bf alo = *(const v8bf*)(abase + k0 + aoff);
    v8bf ahi = *(const v8bf*)(abase + k0 + 8 + aoff);
    v16bf a = __builtin_shufflevector(alo, ahi,
                0,1,2,3,4,5,6,7,8,9,10,11,12,13,14,15);
    v16bf b = *(const v16bf*)(bbase + k0 + bkoff);
    c = __builtin_amdgcn_wmma_f32_16x16x32_bf16(
          false, a, false, b, (short)0, c, false, false);
  }

  int rbase = m0 + ((lane >> 4) << 3);
  int col   = n0 + (lane & 15);
#pragma unroll
  for (int v = 0; v < 8; ++v) xdbl[(rbase + v) * 48 + col] = c[v];
}

// ---------------------------------------------------------------------------
// K6: chunked selective scan. Thread = (b, d, chunk). dt fused inline:
// dt = softplus(x_dbl[:, :16] @ dt_w[d] + dt_b[d]). Emits per-chunk partials:
// sum(dt) (chunk decay = exp(A_n * sum dt)) and chunk-local state h.
// ---------------------------------------------------------------------------
__global__ void k_scan_chunks(const float* __restrict__ xdbl,
                              const __bf16* __restrict__ xc_bf,
                              const float* __restrict__ dt_w,
                              const float* __restrict__ dt_b,
                              const float* __restrict__ A_log,
                              float* __restrict__ sdt_buf,
                              float* __restrict__ hpart) {
  int tid  = blockIdx.x * blockDim.x + threadIdx.x;  // 8*512*64
  int d    = tid & 511;
  int rest = tid >> 9;
  int b    = rest & 7;
  int c    = rest >> 3;

  float dtw[DTR];
#pragma unroll
  for (int r = 0; r < DTR; ++r) dtw[r] = dt_w[d * DTR + r];
  float dtb = dt_b[d];
  float An[DS];
#pragma unroll
  for (int n = 0; n < DS; ++n) An[n] = -__expf(A_log[d * DS + n]);

  float h[DS];
#pragma unroll
  for (int n = 0; n < DS; ++n) h[n] = 0.f;
  float sdt = 0.f;

  int mbase = b * L_ + c * SCHUNK;
  for (int s = 0; s < SCHUNK; ++s) {
    int m = mbase + s;
    const float* row = xdbl + m * 48;
    float pre = dtb;
#pragma unroll
    for (int r = 0; r < DTR; ++r) pre += row[r] * dtw[r];
    float dt = (pre > 20.f) ? pre : __logf(1.f + __expf(pre));
    sdt += dt;
    float xv  = (float)xc_bf[m * DI + d];
    float dtx = dt * xv;
    const float* Bt = row + DTR;
#pragma unroll
    for (int n = 0; n < DS; ++n)
      h[n] = __expf(dt * An[n]) * h[n] + dtx * Bt[n];
  }

  int bd = b * DI + d;
  sdt_buf[bd * NCH + c] = sdt;
  float* hp = hpart + (bd * NCH + c) * DS;
#pragma unroll
  for (int n = 0; n < DS; ++n) hp[n] = h[n];
}

// ---------------------------------------------------------------------------
// K7: sequential combine over 64 chunks + finalize last-timestep y:
// y = (h_final . C_last + D*xc_last) * silu(z_last)
// ---------------------------------------------------------------------------
__global__ void k_combine(const float* __restrict__ sdt_buf,
                          const float* __restrict__ hpart,
                          const float* __restrict__ A_log,
                          const float* __restrict__ xdbl,
                          const __bf16* __restrict__ xc_bf,
                          const float* __restrict__ Dv,
                          const float* __restrict__ z_last,
                          float* __restrict__ y_fin) {
  int bd = blockIdx.x * blockDim.x + threadIdx.x;  // 4096
  int d = bd & 511, b = bd >> 9;

  float An[DS];
#pragma unroll
  for (int n = 0; n < DS; ++n) An[n] = -__expf(A_log[d * DS + n]);

  float h[DS];
#pragma unroll
  for (int n = 0; n < DS; ++n) h[n] = 0.f;

  for (int c = 0; c < NCH; ++c) {
    float sdt = sdt_buf[bd * NCH + c];
    const float* hp = hpart + (bd * NCH + c) * DS;
#pragma unroll
    for (int n = 0; n < DS; ++n)
      h[n] = __expf(An[n] * sdt) * h[n] + hp[n];
  }

  int mlast = b * L_ + (L_ - 1);
  const float* Ct = xdbl + mlast * 48 + DTR + DS;
  float y = 0.f;
#pragma unroll
  for (int n = 0; n < DS; ++n) y += h[n] * Ct[n];
  y += Dv[d] * (float)xc_bf[mlast * DI + d];
  float z = z_last[bd];
  y *= z / (1.f + __expf(-z));
  y_fin[bd] = y;
}

// ---------------------------------------------------------------------------
// K8: out[b,j] = sum_d y_fin[b,d] * out_w[j,d]   (8x256, K=512)
// ---------------------------------------------------------------------------
__global__ void k_outproj(const float* __restrict__ y_fin,
                          const float* __restrict__ out_w,
                          float* __restrict__ out) {
  int i = blockIdx.x * blockDim.x + threadIdx.x;  // 2048: b*256 + j
  int j = i & 255, b = i >> 8;
  const float* yv = y_fin + b * DI;
  const float* w  = out_w + j * DI;
  float acc = 0.f;
  for (int d = 0; d < DI; ++d) acc += yv[d] * w[d];
  out[i] = acc;
}

// ---------------------------------------------------------------------------
extern "C" void kernel_launch(void* const* d_in, const int* in_sizes, int n_in,
                              void* d_out, int out_size, void* d_ws, size_t ws_size,
                              hipStream_t stream) {
  const float* x      = (const float*)d_in[0];
  const float* ip_w   = (const float*)d_in[1];
  const float* ip_b   = (const float*)d_in[2];
  const float* in_w   = (const float*)d_in[3];
  const float* conv_w = (const float*)d_in[4];
  const float* conv_b = (const float*)d_in[5];
  const float* xp_w   = (const float*)d_in[6];
  const float* dt_w   = (const float*)d_in[7];
  const float* dt_b   = (const float*)d_in[8];
  const float* A_log  = (const float*)d_in[9];
  const float* Dv     = (const float*)d_in[10];
  const float* out_w  = (const float*)d_in[11];
  float* out = (float*)d_out;
  (void)in_sizes; (void)n_in; (void)out_size; (void)ws_size;

  // Workspace layout (~142 MB), 256B-aligned regions.
  char* ws = (char*)d_ws;
  size_t off = 0;
  auto alloc = [&](size_t bytes) -> char* {
    char* p = ws + off;
    off += (bytes + 255) & ~(size_t)255;
    return p;
  };
  __bf16* hbf   = (__bf16*)alloc((size_t)M_ * DM * 2);        // 16 MB
  __bf16* inwbf = (__bf16*)alloc((size_t)1024 * 256 * 2);     // 0.5 MB
  __bf16* xpwbf = (__bf16*)alloc((size_t)48 * 512 * 2);       // 48 KB
  float*  xi    = (float*) alloc((size_t)M_ * DI * 4);        // 64 MB
  __bf16* xcbf  = (__bf16*)alloc((size_t)M_ * DI * 2);        // 32 MB
  float*  xdbl  = (float*) alloc((size_t)M_ * 48 * 4);        // 6 MB
  float*  zlast = (float*) alloc((size_t)B_ * DI * 4);
  float*  sdtb  = (float*) alloc((size_t)B_ * DI * NCH * 4);  // 1 MB
  float*  hpart = (float*) alloc((size_t)B_ * DI * NCH * DS * 4); // 16 MB
  float*  yfin  = (float*) alloc((size_t)B_ * DI * 4);

  // K0: weight conversion (286720 elems)
  k_convert_weights<<<1120, 256, 0, stream>>>(in_w, xp_w, inwbf, xpwbf);
  // K1: input projection (M_*256 threads)
  k_inproj<<<(M_ * DM) / 256, 256, 0, stream>>>(x, ip_w, ip_b, hbf);
  // K2: big GEMM, one M-tile per block (LDS-staged A), 8 waves x 8 N-tiles
  k_gemm_xz<<<M_ / 16, 256, 0, stream>>>(hbf, inwbf, xi, zlast);
  // K3: depthwise conv + silu (M_*512 threads)
  k_conv_silu<<<(M_ * DI) / 256, 256, 0, stream>>>(xi, conv_w, conv_b, xcbf);
  // K4: x_dbl GEMM, 2048*3 wave-tiles
  k_gemm_xdbl<<<(2048 * 3) / 8, 256, 0, stream>>>(xcbf, xpwbf, xdbl);
  // K6: chunked scan (8*512*64 threads)
  k_scan_chunks<<<(B_ * DI * NCH) / 256, 256, 0, stream>>>(
      xdbl, xcbf, dt_w, dt_b, A_log, sdtb, hpart);
  // K7: combine chunks + finalize last-step y (4096 threads)
  k_combine<<<(B_ * DI) / 256, 256, 0, stream>>>(
      sdtb, hpart, A_log, xdbl, xcbf, Dv, zlast, yfin);
  // K8: output projection (2048 threads)
  k_outproj<<<(B_ * DM) / 256, 256, 0, stream>>>(yfin, out_w, out);
}
